// CNNLSTMClassifier_80393197847102
// MI455X (gfx1250) — matrix operand order repaired and checked by reading, not compiled, not run on hardware
//
#include <hip/hip_runtime.h>
#include <hip/hip_bf16.h>
#include <stdint.h>

// ---------------------------------------------------------------------------
// CNN (k=3,4,5) + suffix-max + BiLSTM + attention + FC for MI455X (gfx1250).
// All GEMMs use v_wmma_f32_16x16x32_bf16 (wave32 fragment layouts per CDNA5
// ISA 7.12.2). bf16 operands, f32 accumulation.
// ---------------------------------------------------------------------------

typedef __attribute__((ext_vector_type(16))) __bf16          v16bf;
typedef __attribute__((ext_vector_type(8)))  float           v8f;
typedef __attribute__((ext_vector_type(8)))  unsigned short  ush8;   // 16B -> b128

union FragU { v16bf v; ush8 h[2]; };

__device__ __forceinline__ unsigned short f2bf(float x) {
  unsigned u = __float_as_uint(x);
  u += 0x7FFFu + ((u >> 16) & 1u);          // round-to-nearest-even
  return (unsigned short)(u >> 16);
}
__device__ __forceinline__ float sigm(float x) { return 1.0f / (1.0f + __expf(-x)); }

#define BB 16
#define SS 512
#define EE 256
#define FF 128
#define HH 256
#define DD 384
#define TT 508
#define GG 1024

// ---------------- embedding gather + bf16 cast (time-major [B][S][E]) -------
__global__ void k_embed(const int* __restrict__ ids, const float* __restrict__ table,
                        unsigned short* __restrict__ xbf) {
  const int row = blockIdx.x;                 // b*512 + s
  const int e   = threadIdx.x;                // 0..255
  const int id  = ids[row];
  const float v = (id == 0) ? 0.0f : table[(long)id * EE + e];
  xbf[(long)row * EE + e] = f2bf(v);
}

// ------------- conv weight reorder: dst[f][j*E+e] = w[f][e][j], bf16 --------
__global__ void k_convw(const float* __restrict__ w, unsigned short* __restrict__ dst, int KW) {
  const int idx = blockIdx.x * blockDim.x + threadIdx.x;
  const int tot = FF * EE * KW;
  if (idx >= tot) return;
  const int f   = idx / (EE * KW);
  const int rem = idx % (EE * KW);
  const int j   = rem / EE;
  const int e   = rem % EE;
  dst[idx] = f2bf(w[(long)f * EE * KW + (long)e * KW + j]);
}

// ------------------------------- f32 -> bf16 copy ---------------------------
__global__ void k_cast(const float* __restrict__ s, unsigned short* __restrict__ d, int n) {
  const int i = blockIdx.x * blockDim.x + threadIdx.x;
  if (i < n) d[i] = f2bf(s[i]);
}

// ------------------------------- b_ih + b_hh --------------------------------
__global__ void k_bias(const float* __restrict__ a, const float* __restrict__ b,
                       float* __restrict__ o) {
  const int i = blockIdx.x * blockDim.x + threadIdx.x;
  if (i < GG) o[i] = a[i] + b[i];
}

// --------- conv as WMMA GEMM: out[t][f] = relu(patch[t][:] @ Wc[f][:].T) ----
// grid: (32 time-tiles, B). block: 256 = 8 waves, wave -> 16-wide f tile.
// co is padded to 512 time rows; rows >= Sp hold garbage and are never read,
// so the store epilogue is unconditional (no exec-mask churn).
__global__ void __launch_bounds__(256)
k_conv(const unsigned short* __restrict__ xbf, const unsigned short* __restrict__ wc,
       const float* __restrict__ bias, float* __restrict__ co, int KW) {
  const int b    = blockIdx.y;
  const int t0   = blockIdx.x * 16;
  const int wave = threadIdx.x >> 5;
  const int lane = threadIdx.x & 31;
  const int lo   = (lane < 16);
  const int l15  = lane & 15;
  const int f0   = wave * 16;
  const int Ktot = KW * EE;

  const unsigned short* bptr = wc + (long)(f0 + l15) * Ktot + (lo ? 0 : 8);
  v8f acc = {};
  for (int k0 = 0; k0 < Ktot; k0 += 32) {
    const int j  = k0 >> 8;                    // kk = j*E + e, E=256
    const int e0 = k0 & 255;
    int t = t0 + l15 + j; if (t > SS - 1) t = SS - 1;   // clamp OOB rows (discarded)
    const unsigned short* ap = xbf + ((long)(b * SS + t)) * EE + e0 + (lo ? 0 : 8);
    FragU A, Bm;
    A.h[0]  = *(const ush8*)(ap);
    A.h[1]  = *(const ush8*)(ap + 16);
    Bm.h[0] = *(const ush8*)(bptr + k0);
    Bm.h[1] = *(const ush8*)(bptr + k0 + 16);
    acc = __builtin_amdgcn_wmma_f32_16x16x32_bf16(false, A.v, false, Bm.v,
                                                  (short)0, acc, false, false);
  }
  const float bs = bias[f0 + l15];
  float* cp = co + ((long)b * SS + t0 + (lo ? 0 : 8)) * FF + f0 + l15;
#pragma unroll
  for (int r = 0; r < 8; ++r) {
    const float v = acc[r] + bs;
    cp[(long)r * FF] = v > 0.f ? v : 0.f;      // unconditional, coalesced over l15
  }
}

// --------- suffix max over time + concat into cnn[b][t][ci*F+f] bf16 --------
__global__ void k_sufmax(const float* __restrict__ co, unsigned short* __restrict__ cnn) {
  const int idx = blockIdx.x * blockDim.x + threadIdx.x;  // (ci,b,f)
  if (idx >= 3 * BB * FF) return;
  const int f  = idx & 127;
  const int b  = (idx >> 7) & 15;
  const int ci = idx >> 11;
  const int Sp = SS - (ci + 3) + 1;                       // 510,509,508
  const float* src = co + (((long)ci * BB + b) * SS) * FF + f;
  unsigned short* dst = cnn + ((long)b * TT) * DD + ci * FF + f;
  float m = -3.4e38f;
  for (int t = Sp - 1; t >= 0; --t) {
    const float v = src[(long)t * FF];
    m = v > m ? v : m;
    if (t < TT) dst[(long)t * DD] = f2bf(m);
  }
}

// ------ input projections: pre[dir][b*T+t][g] = cnn[b][t][:] @ w_ih.T -------
// grid: (508 m-tiles, 8 gate groups, 2 dirs). block 256 = 8 waves.
// Output kept in the GEMM's natural row order (m = b*T+t) so the epilogue is
// a dense store with no integer division.
__global__ void __launch_bounds__(256)
k_inproj(const unsigned short* __restrict__ cnn, const unsigned short* __restrict__ wih,
         float* __restrict__ pre) {
  const int mt   = blockIdx.x;
  const int dir  = blockIdx.z;
  const int wave = threadIdx.x >> 5;
  const int lane = threadIdx.x & 31;
  const int lo   = (lane < 16);
  const int l15  = lane & 15;
  const int g0   = blockIdx.y * 128 + wave * 16;

  const unsigned short* wptr = wih + (long)dir * GG * DD + (long)(g0 + l15) * DD + (lo ? 0 : 8);
  const unsigned short* aptr = cnn + (long)(mt * 16 + l15) * DD + (lo ? 0 : 8);
  float* po = pre + (long)dir * TT * BB * GG;

  v8f acc = {};
#pragma unroll
  for (int k0 = 0; k0 < DD; k0 += 32) {
    FragU A, Bm;
    A.h[0]  = *(const ush8*)(aptr + k0);
    A.h[1]  = *(const ush8*)(aptr + k0 + 16);
    Bm.h[0] = *(const ush8*)(wptr + k0);
    Bm.h[1] = *(const ush8*)(wptr + k0 + 16);
    acc = __builtin_amdgcn_wmma_f32_16x16x32_bf16(false, A.v, false, Bm.v,
                                                  (short)0, acc, false, false);
  }
  float* op = po + ((long)(mt * 16 + (lo ? 0 : 8))) * GG + g0 + l15;
#pragma unroll
  for (int r = 0; r < 8; ++r) op[(long)r * GG] = acc[r];
}

// ------------- LSTM recurrence: one persistent workgroup per direction ------
// 512 threads = 16 waves; wave owns 64 gate columns (4 WMMA N-tiles).
// h (bf16) and gate pre-activations exchanged through LDS; c stays in regs.
// pre is laid out [dir][b*T+t][g] -> batch stride T*G, no division needed.
__global__ void __launch_bounds__(512)
k_lstm(const float* __restrict__ pre, const unsigned short* __restrict__ whh,
       const float* __restrict__ bsum, float* __restrict__ hs) {
  const int dir = blockIdx.x;
  __shared__ unsigned short hbf[BB * HH];   // 8 KB  : h[b][h] bf16 (A-matrix)
  __shared__ float          gl[BB * GG];    // 64 KB : gate pre-activations

  const int tid  = threadIdx.x;
  const int wave = tid >> 5;
  const int lane = tid & 31;
  const int lo   = (lane < 16);
  const int l15  = lane & 15;
  const int cb   = tid >> 5;                // cell batch (0..15)
  const int ch0  = (tid & 31) * 8;          // 8 cells per thread

  for (int i = tid; i < BB * HH; i += 512) hbf[i] = 0;
  float creg[8];
#pragma unroll
  for (int i = 0; i < 8; ++i) creg[i] = 0.f;

  const float*          pred = pre  + (long)dir * TT * BB * GG;
  const unsigned short* whd  = whh  + (long)dir * GG * HH;
  const float*          bsd  = bsum + dir * GG;
  float*                hsd  = hs   + (long)dir * TT * BB * HH;

  __syncthreads();
  for (int step = 0; step < TT; ++step) {
    const int t = (dir == 0) ? step : (TT - 1 - step);
    // ---- g = h_prev @ W_hh.T  (M=16, N=1024, K=256) ----
    for (int nt = 0; nt < 4; ++nt) {
      const int g0 = wave * 64 + nt * 16;
      const unsigned short* wptr = whd + (long)(g0 + l15) * HH + (lo ? 0 : 8);
      const unsigned short* aptr = hbf + l15 * HH + (lo ? 0 : 8);
      v8f acc = {};
#pragma unroll
      for (int k0 = 0; k0 < HH; k0 += 32) {
        FragU A, Bm;
        A.h[0]  = *(const ush8*)(aptr + k0);
        A.h[1]  = *(const ush8*)(aptr + k0 + 16);
        Bm.h[0] = *(const ush8*)(wptr + k0);
        Bm.h[1] = *(const ush8*)(wptr + k0 + 16);
        acc = __builtin_amdgcn_wmma_f32_16x16x32_bf16(false, A.v, false, Bm.v,
                                                      (short)0, acc, false, false);
      }
      const float* pp  = pred + (long)t * GG + g0 + l15;      // + m*T*G per row
      const float  bsv = bsd[g0 + l15];
#pragma unroll
      for (int r = 0; r < 8; ++r) {
        const int m = r + (lo ? 0 : 8);     // batch row
        gl[m * GG + g0 + l15] = acc[r] + pp[(long)m * TT * GG] + bsv;
      }
    }
    __syncthreads();                         // hbf reads + gl writes done
    // ---- cell update (i,f,g,o in PyTorch order) ----
#pragma unroll
    for (int i = 0; i < 8; ++i) {
      const int h  = ch0 + i;
      const float gi = gl[cb * GG + h];
      const float gf = gl[cb * GG + 256 + h];
      const float gg = gl[cb * GG + 512 + h];
      const float go = gl[cb * GG + 768 + h];
      const float c  = sigm(gf) * creg[i] + sigm(gi) * tanhf(gg);
      creg[i] = c;
      const float hh = sigm(go) * tanhf(c);
      hsd[((long)t * BB + cb) * HH + h] = hh;
      hbf[cb * HH + h] = f2bf(hh);
    }
    __syncthreads();                         // new hbf visible to all waves
  }
}

// ------------------- attention softmax + pooling + FC -----------------------
__global__ void __launch_bounds__(512)
k_att(const float* __restrict__ hs, const float* __restrict__ attw,
      const float* __restrict__ attb, const float* __restrict__ fcw,
      const float* __restrict__ fcb, float* __restrict__ out) {
  const int b   = blockIdx.x;
  const int tid = threadIdx.x;
  __shared__ float sc[512];
  __shared__ float red[512];
  __shared__ float pooled[512];
  const float* hf = hs;
  const float* hb = hs + (long)TT * BB * HH;

  float s = -3.4e38f;
  if (tid < TT) {
    float acc = attb[0];
    const float* pf = hf + ((long)tid * BB + b) * HH;
    const float* pb = hb + ((long)tid * BB + b) * HH;
    for (int h = 0; h < HH; ++h) acc += pf[h] * attw[h] + pb[h] * attw[HH + h];
    s = acc;
  }
  red[tid] = s;
  __syncthreads();
  for (int off = 256; off > 0; off >>= 1) {
    if (tid < off) { const float o = red[tid + off]; if (o > red[tid]) red[tid] = o; }
    __syncthreads();
  }
  const float mx = red[0];
  __syncthreads();
  const float e = (tid < TT) ? __expf(s - mx) : 0.f;
  sc[tid]  = e;
  red[tid] = e;
  __syncthreads();
  for (int off = 256; off > 0; off >>= 1) {
    if (tid < off) red[tid] += red[tid + off];
    __syncthreads();
  }
  const float denom = red[0];
  __syncthreads();
  {
    const int hc = tid;
    const float* src = (hc < HH) ? (hf + (long)b * HH + hc)
                                 : (hb + (long)b * HH + (hc - HH));
    float acc = 0.f;
    for (int t = 0; t < TT; ++t) acc += sc[t] * src[(long)t * BB * HH];
    pooled[hc] = acc / denom;
  }
  __syncthreads();
  if (tid < 8) {
    float acc = fcb[tid];
    for (int hc = 0; hc < 512; ++hc) acc += pooled[hc] * fcw[tid * 512 + hc];
    out[b * 8 + tid] = acc;
  }
}

// ---------------------------------------------------------------------------
extern "C" void kernel_launch(void* const* d_in, const int* in_sizes, int n_in,
                              void* d_out, int out_size, void* d_ws, size_t ws_size,
                              hipStream_t stream) {
  (void)in_sizes; (void)n_in; (void)out_size; (void)ws_size;
  const int*   ids    = (const int*)  d_in[0];
  const float* table  = (const float*)d_in[1];
  const float* cw3    = (const float*)d_in[2];
  const float* cb3    = (const float*)d_in[3];
  const float* cw4    = (const float*)d_in[4];
  const float* cb4    = (const float*)d_in[5];
  const float* cw5    = (const float*)d_in[6];
  const float* cb5    = (const float*)d_in[7];
  const float* wihf   = (const float*)d_in[8];
  const float* whhf   = (const float*)d_in[9];
  const float* bihf   = (const float*)d_in[10];
  const float* bhhf   = (const float*)d_in[11];
  const float* wihb   = (const float*)d_in[12];
  const float* whhb   = (const float*)d_in[13];
  const float* bihb   = (const float*)d_in[14];
  const float* bhhb   = (const float*)d_in[15];
  const float* attw   = (const float*)d_in[16];
  const float* attb   = (const float*)d_in[17];
  const float* fcw    = (const float*)d_in[18];
  const float* fcb    = (const float*)d_in[19];
  float* out = (float*)d_out;

  uint8_t* wp = (uint8_t*)d_ws;
  auto alloc = [&](size_t bytes) {
    uint8_t* p = wp;
    wp += (bytes + 255) & ~(size_t)255;
    return p;
  };
  unsigned short* xbf  = (unsigned short*)alloc((size_t)BB * SS * EE * 2);
  unsigned short* wc3  = (unsigned short*)alloc((size_t)FF * EE * 3 * 2);
  unsigned short* wc4  = (unsigned short*)alloc((size_t)FF * EE * 4 * 2);
  unsigned short* wc5  = (unsigned short*)alloc((size_t)FF * EE * 5 * 2);
  unsigned short* wih  = (unsigned short*)alloc((size_t)2 * GG * DD * 2);
  unsigned short* whh  = (unsigned short*)alloc((size_t)2 * GG * HH * 2);
  float*          bsum = (float*)         alloc((size_t)2 * GG * 4);
  float*          co   = (float*)         alloc((size_t)3 * BB * SS * FF * 4);
  unsigned short* cnn  = (unsigned short*)alloc((size_t)BB * TT * DD * 2);
  float*          pre  = (float*)         alloc((size_t)2 * TT * BB * GG * 4);
  float*          hsb  = (float*)         alloc((size_t)2 * TT * BB * HH * 4);

  // --- stage 0: casts / reorders ---
  k_embed<<<BB * SS, EE, 0, stream>>>(ids, table, xbf);
  k_convw<<<(FF * EE * 3 + 255) / 256, 256, 0, stream>>>(cw3, wc3, 3);
  k_convw<<<(FF * EE * 4 + 255) / 256, 256, 0, stream>>>(cw4, wc4, 4);
  k_convw<<<(FF * EE * 5 + 255) / 256, 256, 0, stream>>>(cw5, wc5, 5);
  k_cast<<<(GG * DD + 255) / 256, 256, 0, stream>>>(wihf, wih, GG * DD);
  k_cast<<<(GG * DD + 255) / 256, 256, 0, stream>>>(wihb, wih + (size_t)GG * DD, GG * DD);
  k_cast<<<(GG * HH + 255) / 256, 256, 0, stream>>>(whhf, whh, GG * HH);
  k_cast<<<(GG * HH + 255) / 256, 256, 0, stream>>>(whhb, whh + (size_t)GG * HH, GG * HH);
  k_bias<<<(GG + 255) / 256, 256, 0, stream>>>(bihf, bhhf, bsum);
  k_bias<<<(GG + 255) / 256, 256, 0, stream>>>(bihb, bhhb, bsum + GG);

  // --- stage 1: convs (WMMA) ---
  k_conv<<<dim3(32, BB), 256, 0, stream>>>(xbf, wc3, cb3, co,                            3);
  k_conv<<<dim3(32, BB), 256, 0, stream>>>(xbf, wc4, cb4, co + (size_t)BB * SS * FF,     4);
  k_conv<<<dim3(32, BB), 256, 0, stream>>>(xbf, wc5, cb5, co + (size_t)2 * BB * SS * FF, 5);

  // --- stage 2: suffix max-pool + concat ---
  k_sufmax<<<(3 * BB * FF + 255) / 256, 256, 0, stream>>>(co, cnn);

  // --- stage 3: LSTM input projections (WMMA, both dirs) ---
  k_inproj<<<dim3(TT, 8, 2), 256, 0, stream>>>(cnn, wih, pre);

  // --- stage 4: recurrence (WMMA + LDS-resident state) ---
  k_lstm<<<2, 512, 0, stream>>>(pre, whh, bsum, hsb);

  // --- stage 5: attention + FC ---
  k_att<<<BB, 512, 0, stream>>>(hsb, attw, attb, fcw, fcb, out);
}
